// Mamba_75505525063788
// MI455X (gfx1250) — matrix operand order repaired and checked by reading, not compiled, not run on hardware
//
#include <hip/hip_runtime.h>
#include <math.h>

typedef __bf16 bf16;
typedef __attribute__((ext_vector_type(16))) __bf16 v16bf;
typedef __attribute__((ext_vector_type(8)))  float  v8f;

#define B_   8
#define L_   2049
#define D_   256
#define DI   512
#define DST  16
#define DTR  16
#define DCV  4
#define S_   64
#define MID_ (L_/2 + 1)          /* 1025 */
#define LC_  (MID_ + DCV - 1)    /* 1028: conv out = 1025 + 3 + 3 - 4 + 1 */
#define NROW (B_*LC_)            /* 8224 */
#define NRIN (B_*L_)             /* 16392 */
#define KP_  (((LC_ + 31)/32)*32) /* 1056: K padded for WMMA attn GEMM */

__device__ __forceinline__ float siluf(float x) { return x / (1.0f + expf(-x)); }
__device__ __forceinline__ float softplusf(float x) {
    return (x > 0.0f) ? (x + log1pf(expf(-x))) : log1pf(expf(x));
}

// ---------------------------------------------------------------- converts
__global__ __launch_bounds__(256) void k_cvt_bf16(const float* __restrict__ in,
                                                  bf16* __restrict__ out, int n) {
    int i = blockIdx.x * 256 + threadIdx.x;
    if (i < n) out[i] = (bf16)in[i];
}

// out[c*rows + r] = in[r*cols + c]  (transpose + convert; for token_wV)
__global__ __launch_bounds__(256) void k_cvt_t_bf16(const float* __restrict__ in,
                                                    bf16* __restrict__ out,
                                                    int rows, int cols) {
    int i = blockIdx.x * 256 + threadIdx.x;
    if (i >= rows * cols) return;
    int r = i / cols, c = i % cols;
    out[(size_t)c * rows + r] = (bf16)in[i];
}

// Atok (B,S,LC) f32 -> (B,S,KP) bf16 zero-padded in K
__global__ __launch_bounds__(256) void k_cvt_atok(const float* __restrict__ Atok,
                                                  bf16* __restrict__ out) {
    int i = blockIdx.x * 256 + threadIdx.x;
    if (i >= B_ * S_ * KP_) return;
    const int l = i % KP_;
    const int bt = i / KP_;
    out[i] = (bf16)((l < LC_) ? Atok[(size_t)bt * LC_ + l] : 0.0f);
}

// VV (B,LC,DI) f32 -> VVT (B,DI,KP) bf16 zero-padded in K (weight layout per batch)
__global__ __launch_bounds__(256) void k_cvt_vvt(const float* __restrict__ VV,
                                                 bf16* __restrict__ out) {
    int i = blockIdx.x * 256 + threadIdx.x;
    if (i >= B_ * DI * KP_) return;
    const int l = i % KP_;
    const int be = i / KP_;
    const int e = be % DI, b = be / DI;
    out[i] = (bf16)((l < LC_) ? VV[((size_t)(b * LC_ + l)) * DI + e] : 0.0f);
}

// ---------------------------------------------------------------- WMMA GEMM
// C[M,N] (f32) = A[M,K] (bf16 row-major) * Bw[N,K]^T (bf16 weight layout) + bias
// Each wave owns a 16x64 output strip. Per K-step: load the A fragment once,
// load ALL FOUR B fragments into distinct registers (loads overlap as a
// clause), then issue 4 back-to-back v_wmma_f32_16x16x32_bf16.
__global__ __launch_bounds__(256) void k_wmma_gemm(const bf16* __restrict__ A,
                                                   const bf16* __restrict__ Bw,
                                                   float* __restrict__ C,
                                                   const float* __restrict__ bias,
                                                   int M, int N, int K) {
    const int lane = threadIdx.x & 31;
    const int wid  = threadIdx.x >> 5;
    const int tiles_n = N >> 6;                      // 64-wide N strips
    const int tiles_m = (M + 15) >> 4;
    int tile = blockIdx.x * 8 + wid;
    if (tile >= tiles_m * tiles_n) return;           // wave-uniform exit
    const int tm = tile / tiles_n, tn = tile % tiles_n;
    const int half = lane >> 4, l15 = lane & 15;

    int am = tm * 16 + l15;                          // A row for this lane
    if (am >= M) am = M - 1;                         // clamp (store is guarded)
    const bf16* ap  = A  + (size_t)am * K;
    const bf16* bp0 = Bw + (size_t)(tn * 64 + l15) * K;

    v8f c[4] = {v8f{}, v8f{}, v8f{}, v8f{}};
    for (int k0 = 0; k0 < K; k0 += 32) {
        v16bf a;
        // A 16x32 layout: lanes 0-15 K{0..7,16..23}, lanes 16-31 K{8..15,24..31}
        const int ka  = k0 + (half ? 8  : 0);
        const int ka2 = k0 + (half ? 24 : 16);
#pragma unroll
        for (int v = 0; v < 4; ++v) {
            a[2*v]         = ap[ka  + 2*v];
            a[2*v + 1]     = ap[ka  + 2*v + 1];
            a[8 + 2*v]     = ap[ka2 + 2*v];
            a[8 + 2*v + 1] = ap[ka2 + 2*v + 1];
        }
        // B 32x16 layout: lanes 0-15 K{0..7,16..23}, lanes 16-31 K{8..15,24..31}
        const int kb = k0 + (half ? 8 : 0);
        if (k0 + 32 < K) {
            __builtin_prefetch(ap + k0 + 32, 0, 3);
            __builtin_prefetch(bp0 + k0 + 32, 0, 3);
        }
        // Load all four B fragments first so their global_load_b128s overlap,
        // then run the 4 WMMAs back-to-back.
        v16bf bfr[4];
#pragma unroll
        for (int j = 0; j < 4; ++j) {
            const bf16* bp = bp0 + (size_t)(16 * j) * K;
#pragma unroll
            for (int e = 0; e < 16; ++e) bfr[j][e] = bp[kb + (e < 8 ? e : e + 8)];
        }
#pragma unroll
        for (int j = 0; j < 4; ++j) {
            c[j] = __builtin_amdgcn_wmma_f32_16x16x32_bf16(false, a, false, bfr[j],
                                                           (short)0, c[j], false, false);
        }
    }
    // C/D layout: lane -> N, VGPR r -> M (lanes>=16 add 8)
#pragma unroll
    for (int j = 0; j < 4; ++j) {
        const int ng = tn * 64 + 16 * j + l15;
        const float bv = bias ? bias[ng] : 0.0f;
#pragma unroll
        for (int r = 0; r < 8; ++r) {
            int mg = tm * 16 + (half ? 8 : 0) + r;
            if (mg < M) C[(size_t)mg * N + ng] = c[j][r] + bv;
        }
    }
}

// ---------------------------------------------------------------- conv + silu
// dir 0: forward half xi[:, :1025];  dir 1: backward half xi[:, 2048-i]
__global__ __launch_bounds__(256) void k_conv_silu(const float* __restrict__ xi_z,
                                                   const float* __restrict__ wf,
                                                   const float* __restrict__ bf_,
                                                   const float* __restrict__ wb,
                                                   const float* __restrict__ bb_,
                                                   float* __restrict__ xf,
                                                   float* __restrict__ xb) {
    int idx = blockIdx.x * 256 + threadIdx.x;
    if (idx >= NROW * DI) return;
    const int dir = blockIdx.y;
    const int c = idx % DI;
    const int bt = idx / DI;
    const int t = bt % LC_, b = bt / LC_;
    const float* w    = dir ? wb  : wf;
    const float* bias = dir ? bb_ : bf_;
    float acc = bias[c];
#pragma unroll
    for (int j = 0; j < DCV; ++j) {
        int i = t - 3 + j;
        if (i >= 0 && i < MID_) {
            int l = dir ? (L_ - 1 - i) : i;
            acc += w[c * DCV + j] * xi_z[((size_t)(b * L_ + l)) * (2 * DI) + c];
        }
    }
    float* out = dir ? xb : xf;
    out[(size_t)idx] = siluf(acc);
}

// ---------------------------------------------------------------- x_proj (dbc)
__global__ __launch_bounds__(256) void k_xproj(const float* __restrict__ xf,
                                               const float* __restrict__ xb,
                                               const float* __restrict__ Wf,
                                               const float* __restrict__ bf_,
                                               const float* __restrict__ Wb,
                                               const float* __restrict__ bb_,
                                               float* __restrict__ dbcf,
                                               float* __restrict__ dbcb) {
    int idx = blockIdx.x * 256 + threadIdx.x;
    const int NC = DTR + 2 * DST; /* 48 */
    if (idx >= NROW * NC) return;
    const int dir = blockIdx.y;
    const int n = idx % NC;
    const int row = idx / NC;
    const float* u = dir ? xb : xf;
    const float* W = dir ? Wb : Wf;
    float acc = (dir ? bb_ : bf_)[n];
    const float* up = u + (size_t)row * DI;
    const float* wp = W + (size_t)n * DI;
    for (int k = 0; k < DI; ++k) acc += up[k] * wp[k];
    (dir ? dbcb : dbcf)[(size_t)row * NC + n] = acc;
}

// ---------------------------------------------------------------- delta = softplus(dr@Wdt.T + bdt)
__global__ __launch_bounds__(256) void k_delta(const float* __restrict__ dbcf,
                                               const float* __restrict__ dbcb,
                                               const float* __restrict__ Wf,
                                               const float* __restrict__ bf_,
                                               const float* __restrict__ Wb,
                                               const float* __restrict__ bb_,
                                               float* __restrict__ dltf,
                                               float* __restrict__ dltb) {
    int idx = blockIdx.x * 256 + threadIdx.x;
    if (idx >= NROW * DI) return;
    const int dir = blockIdx.y;
    const int c = idx % DI;
    const int row = idx / DI;
    const float* dbc = dir ? dbcb : dbcf;
    const float* W   = dir ? Wb : Wf;
    float acc = (dir ? bb_ : bf_)[c];
    const float* dp = dbc + (size_t)row * 48;
    const float* wp = W + (size_t)c * DTR;
#pragma unroll
    for (int r = 0; r < DTR; ++r) acc += dp[r] * wp[r];
    (dir ? dltb : dltf)[(size_t)idx] = softplusf(acc);
}

// ---------------------------------------------------------------- SSM scan
// block = (b, dir), 512 threads (one channel each); 16 states in registers.
__global__ __launch_bounds__(512) void k_scan(const float* __restrict__ xf,
                                              const float* __restrict__ xb,
                                              const float* __restrict__ dltf,
                                              const float* __restrict__ dltb,
                                              const float* __restrict__ dbcf,
                                              const float* __restrict__ dbcb,
                                              const float* __restrict__ Alf,
                                              const float* __restrict__ Alb,
                                              const float* __restrict__ Dfv,
                                              const float* __restrict__ Dbv,
                                              float* __restrict__ yf,
                                              float* __restrict__ yb) {
    const int dir = blockIdx.y;
    const int b = blockIdx.x;
    const int c = threadIdx.x;
    const float* u   = dir ? xb   : xf;
    const float* dlt = dir ? dltb : dltf;
    const float* dbc = dir ? dbcb : dbcf;
    const float* Al  = dir ? Alb  : Alf;
    const float  Dc  = (dir ? Dbv : Dfv)[c];
    float* y = dir ? yb : yf;

    float a[DST], h[DST];
#pragma unroll
    for (int s = 0; s < DST; ++s) { a[s] = -expf(Al[c * DST + s]); h[s] = 0.0f; }

    __shared__ float sBC[32];
    for (int t = 0; t < LC_; ++t) {
        const size_t row = (size_t)b * LC_ + t;
        if (threadIdx.x < 32) sBC[threadIdx.x] = dbc[row * 48 + 16 + threadIdx.x];
        __syncthreads();
        const float d  = dlt[row * DI + c];
        const float uu = u[row * DI + c];
        const float du = d * uu;
        float acc = 0.0f;
#pragma unroll
        for (int s = 0; s < DST; ++s) {
            float dA = expf(d * a[s]);
            h[s] = dA * h[s] + du * sBC[s];
            acc += h[s] * sBC[16 + s];
        }
        y[row * DI + c] = acc + Dc * uu;
        __syncthreads();
    }
}

// ---------------------------------------------------------------- mask: row distances
__global__ __launch_bounds__(128) void k_rowdist(const float* __restrict__ y,
                                                 float* __restrict__ dv, int refIdx) {
    const int row = blockIdx.x;                     // 0..NROW-1
    const int b = row / LC_;
    const int tid = threadIdx.x;
    const float* yr = y + (size_t)row * DI;
    const float* yf = y + ((size_t)(b * LC_ + refIdx)) * DI;
    __shared__ float red[128];
    float ss = 0.0f;
    for (int c = tid; c < DI; c += 128) { float df = yr[c] - yf[c]; ss += df * df; }
    red[tid] = ss; __syncthreads();
    for (int off = 64; off > 0; off >>= 1) {
        if (tid < off) red[tid] += red[tid + off];
        __syncthreads();
    }
    if (tid == 0) dv[row] = sqrtf(fmaxf(red[0], 1e-12f));
}

// index-mask * vector-mask, then L2 normalize over t (scalar norms cancel)
__global__ __launch_bounds__(256) void k_mask(const float* __restrict__ dv,
                                              float* __restrict__ m,
                                              float refI, float sigI) {
    const int b = blockIdx.x;
    const int tid = threadIdx.x;
    __shared__ float red[256];
    __shared__ float sv[2];
    float s = 0.0f;
    for (int t = tid; t < LC_; t += 256) s += dv[b * LC_ + t];
    red[tid] = s; __syncthreads();
    for (int off = 128; off > 0; off >>= 1) { if (tid < off) red[tid] += red[tid + off]; __syncthreads(); }
    if (tid == 0) sv[0] = red[0] / (float)LC_;
    __syncthreads();
    const float sigma = sv[0];
    float ss = 0.0f;
    for (int t = tid; t < LC_; t += 256) {
        float dd = dv[b * LC_ + t] / sigma;
        float di = ((float)t - refI) / sigI;
        float w = expf(-0.5f * (dd * dd + di * di));
        m[b * LC_ + t] = w;
        ss += w * w;
    }
    __syncthreads();
    red[tid] = ss; __syncthreads();
    for (int off = 128; off > 0; off >>= 1) { if (tid < off) red[tid] += red[tid + off]; __syncthreads(); }
    if (tid == 0) sv[1] = fmaxf(sqrtf(red[0]), 1e-12f);
    __syncthreads();
    const float inv = 1.0f / sv[1];
    for (int t = tid; t < LC_; t += 256) m[b * LC_ + t] *= inv;
}

// y_cat = [y_f * m_f  |  y_b * reverse(m_b)]  (bf16 for the WMMA projection)
__global__ __launch_bounds__(256) void k_build_ycat(const float* __restrict__ yf,
                                                    const float* __restrict__ yb,
                                                    const float* __restrict__ mf,
                                                    const float* __restrict__ mb,
                                                    bf16* __restrict__ ycat) {
    int idx = blockIdx.x * 256 + threadIdx.x;
    if (idx >= NROW * DI) return;
    const int c = idx % DI;
    const int row = idx / DI;
    const int t = row % LC_, b = row / LC_;
    const int rrow = b * LC_ + (LC_ - 1 - t);
    ycat[(size_t)row * (2 * DI) + c]      = (bf16)(yf[(size_t)row * DI + c] * mf[row]);
    ycat[(size_t)row * (2 * DI) + DI + c] = (bf16)(yb[(size_t)row * DI + c] * mb[rrow]);
}

__global__ __launch_bounds__(256) void k_apply_mask_cvt(const float* __restrict__ y2,
                                                        const float* __restrict__ mc,
                                                        bf16* __restrict__ out) {
    int idx = blockIdx.x * 256 + threadIdx.x;
    if (idx >= NROW * DI) return;
    const int row = idx / DI;
    out[idx] = (bf16)(y2[idx] * mc[row]);
}

// softmax over l for each (b, token); logits stored [(b*LC+l)*64 + token]
__global__ __launch_bounds__(256) void k_softmax(const float* __restrict__ logits,
                                                 float* __restrict__ Atok) {
    const int bt = blockIdx.x;                      // b*64 + token
    const int b = bt / S_, tk = bt % S_;
    const int tid = threadIdx.x;
    __shared__ float red[256];
    __shared__ float sv[2];
    float mx = -3.4e38f;
    for (int l = tid; l < LC_; l += 256)
        mx = fmaxf(mx, logits[((size_t)(b * LC_ + l)) * S_ + tk]);
    red[tid] = mx; __syncthreads();
    for (int off = 128; off > 0; off >>= 1) { if (tid < off) red[tid] = fmaxf(red[tid], red[tid + off]); __syncthreads(); }
    if (tid == 0) sv[0] = red[0];
    __syncthreads();
    mx = sv[0];
    float s = 0.0f;
    for (int l = tid; l < LC_; l += 256)
        s += expf(logits[((size_t)(b * LC_ + l)) * S_ + tk] - mx);
    __syncthreads();
    red[tid] = s; __syncthreads();
    for (int off = 128; off > 0; off >>= 1) { if (tid < off) red[tid] += red[tid + off]; __syncthreads(); }
    if (tid == 0) sv[1] = red[0];
    __syncthreads();
    const float inv = 1.0f / sv[1];
    for (int l = tid; l < LC_; l += 256)
        Atok[(size_t)bt * LC_ + l] = expf(logits[((size_t)(b * LC_ + l)) * S_ + tk] - mx) * inv;
}

// zp[b,s,d] = silu(mean over pool segment of z);  z = xi_z[:, :, 512:1024]
__global__ __launch_bounds__(256) void k_pool(const float* __restrict__ xi_z,
                                              float* __restrict__ zp) {
    int idx = blockIdx.x * 256 + threadIdx.x;
    if (idx >= B_ * S_ * DI) return;
    const int d = idx % DI;
    const int bs = idx / DI;
    const int b = bs / S_, s = bs % S_;
    const int s0 = (s * L_) / S_;
    const int s1 = ((s + 1) * L_ + S_ - 1) / S_;
    float acc = 0.0f;
    for (int l = s0; l < s1; ++l)
        acc += xi_z[((size_t)(b * L_ + l)) * (2 * DI) + DI + d];
    acc /= (float)(s1 - s0);
    zp[idx] = siluf(acc);
}

// out[b,s,o] = sum_e (T*zp)[b,s,e] * W_out[o,e]
__global__ __launch_bounds__(256) void k_out(const float* __restrict__ T,
                                             const float* __restrict__ zp,
                                             const float* __restrict__ Wout,
                                             float* __restrict__ out) {
    int idx = blockIdx.x * 256 + threadIdx.x;
    if (idx >= B_ * S_ * D_) return;
    const int o = idx % D_;
    const int bs = idx / D_;
    float acc = 0.0f;
    const float* tp = T + (size_t)bs * DI;
    const float* zpp = zp + (size_t)bs * DI;
    const float* wp = Wout + (size_t)o * DI;
    for (int e = 0; e < DI; ++e) acc += tp[e] * zpp[e] * wp[e];
    out[idx] = acc;
}

// ================================================================ launch
extern "C" void kernel_launch(void* const* d_in, const int* in_sizes, int n_in,
                              void* d_out, int out_size, void* d_ws, size_t ws_size,
                              hipStream_t stream) {
    const float* x        = (const float*)d_in[0];
    const float* W_in_x   = (const float*)d_in[1];
    const float* W_in_z   = (const float*)d_in[2];
    const float* conv_w_f = (const float*)d_in[3];
    const float* conv_b_f = (const float*)d_in[4];
    const float* conv_w_b = (const float*)d_in[5];
    const float* conv_b_b = (const float*)d_in[6];
    const float* W_xp_f   = (const float*)d_in[7];
    const float* b_xp_f   = (const float*)d_in[8];
    const float* W_dt_f   = (const float*)d_in[9];
    const float* b_dt_f   = (const float*)d_in[10];
    const float* A_log_f  = (const float*)d_in[11];
    const float* D_f      = (const float*)d_in[12];
    const float* W_xp_b   = (const float*)d_in[13];
    const float* b_xp_b   = (const float*)d_in[14];
    const float* W_dt_b   = (const float*)d_in[15];
    const float* b_dt_b   = (const float*)d_in[16];
    const float* A_log_b  = (const float*)d_in[17];
    const float* D_b      = (const float*)d_in[18];
    const float* W_pro_to = (const float*)d_in[19];
    const float* b_pro_to = (const float*)d_in[20];
    const float* token_wA = (const float*)d_in[21];
    const float* token_wV = (const float*)d_in[22];
    const float* W_out    = (const float*)d_in[23];
    float* out = (float*)d_out;

    // host-side scalar constants (deterministic CPU math)
    const int refC = (LC_ + 1) / 2;                  // 514
    double aL = 0.0, aC = 0.0;
    for (int i = 0; i < LC_; ++i) {
        aL += fabs((double)i - (double)(LC_ - 1));
        aC += fabs((double)i - (double)refC);
    }
    const float sigLast   = (float)(aL / LC_);
    const float sigCenter = (float)(aC / LC_);

    // bump allocator over workspace
    char* p = (char*)d_ws;
    auto alloc = [&](size_t bytes) -> void* {
        void* r = (void*)p;
        p += (bytes + 255) & ~(size_t)255;
        return r;
    };
    bf16*  x_bf    = (bf16*)alloc((size_t)NRIN * D_ * 2);
    bf16*  Wxz_bf  = (bf16*)alloc((size_t)2 * DI * D_ * 2);
    float* xi_z    = (float*)alloc((size_t)NRIN * 2 * DI * 4);
    float* x_f     = (float*)alloc((size_t)NROW * DI * 4);
    float* x_b     = (float*)alloc((size_t)NROW * DI * 4);
    float* dbc_f   = (float*)alloc((size_t)NROW * 48 * 4);
    float* dbc_b   = (float*)alloc((size_t)NROW * 48 * 4);
    float* dlt_f   = (float*)alloc((size_t)NROW * DI * 4);
    float* dlt_b   = (float*)alloc((size_t)NROW * DI * 4);
    float* y_f     = (float*)alloc((size_t)NROW * DI * 4);
    float* y_b     = (float*)alloc((size_t)NROW * DI * 4);
    float* dv      = (float*)alloc((size_t)NROW * 4);
    float* m_f     = (float*)alloc((size_t)NROW * 4);
    float* m_b     = (float*)alloc((size_t)NROW * 4);
    float* m_c     = (float*)alloc((size_t)NROW * 4);
    bf16*  ycat_bf = (bf16*)alloc((size_t)NROW * 2 * DI * 2);
    bf16*  Wpro_bf = (bf16*)alloc((size_t)DI * 2 * DI * 2);
    float* y2      = (float*)alloc((size_t)NROW * DI * 4);
    bf16*  y2m_bf  = (bf16*)alloc((size_t)NROW * DI * 2);
    bf16*  wA_bf   = (bf16*)alloc((size_t)S_ * DI * 2);
    bf16*  wVT_bf  = (bf16*)alloc((size_t)DI * DI * 2);
    float* logits  = (float*)alloc((size_t)NROW * S_ * 4);
    float* Atok    = (float*)alloc((size_t)B_ * S_ * LC_ * 4);
    float* VV      = (float*)alloc((size_t)NROW * DI * 4);
    bf16*  Atok_bf = (bf16*)alloc((size_t)B_ * S_ * KP_ * 2);
    bf16*  VVT_bf  = (bf16*)alloc((size_t)B_ * DI * KP_ * 2);
    float* Tb      = (float*)alloc((size_t)B_ * S_ * DI * 4);
    float* zp      = (float*)alloc((size_t)B_ * S_ * DI * 4);

    auto cdiv = [](int a, int b) { return (a + b - 1) / b; };
    auto gemm = [&](const bf16* A, const bf16* Bw, float* C, const float* bias,
                    int M, int N, int K) {
        int tiles = cdiv(M, 16) * (N / 64);
        k_wmma_gemm<<<cdiv(tiles, 8), 256, 0, stream>>>(A, Bw, C, bias, M, N, K);
    };

    // 1) in-projection (x -> [xi | z]) via WMMA bf16
    k_cvt_bf16<<<cdiv(NRIN * D_, 256), 256, 0, stream>>>(x, x_bf, NRIN * D_);
    k_cvt_bf16<<<cdiv(DI * D_, 256), 256, 0, stream>>>(W_in_x, Wxz_bf, DI * D_);
    k_cvt_bf16<<<cdiv(DI * D_, 256), 256, 0, stream>>>(W_in_z, Wxz_bf + (size_t)DI * D_, DI * D_);
    gemm(x_bf, Wxz_bf, xi_z, nullptr, NRIN, 2 * DI, D_);

    // 2) depthwise conv + silu (both directions)
    k_conv_silu<<<dim3(cdiv(NROW * DI, 256), 2), 256, 0, stream>>>(
        xi_z, conv_w_f, conv_b_f, conv_w_b, conv_b_b, x_f, x_b);

    // 3) x_proj -> dbc ; dt-proj + softplus -> delta ; SSM scan -> y
    k_xproj<<<dim3(cdiv(NROW * 48, 256), 2), 256, 0, stream>>>(
        x_f, x_b, W_xp_f, b_xp_f, W_xp_b, b_xp_b, dbc_f, dbc_b);
    k_delta<<<dim3(cdiv(NROW * DI, 256), 2), 256, 0, stream>>>(
        dbc_f, dbc_b, W_dt_f, b_dt_f, W_dt_b, b_dt_b, dlt_f, dlt_b);
    k_scan<<<dim3(B_, 2), DI, 0, stream>>>(
        x_f, x_b, dlt_f, dlt_b, dbc_f, dbc_b, A_log_f, A_log_b, D_f, D_b, y_f, y_b);

    // 4) masks (kind='last') and concat
    k_rowdist<<<NROW, 128, 0, stream>>>(y_f, dv, LC_ - 1);
    k_mask<<<B_, 256, 0, stream>>>(dv, m_f, (float)(LC_ - 1), sigLast);
    k_rowdist<<<NROW, 128, 0, stream>>>(y_b, dv, LC_ - 1);
    k_mask<<<B_, 256, 0, stream>>>(dv, m_b, (float)(LC_ - 1), sigLast);
    k_build_ycat<<<cdiv(NROW * DI, 256), 256, 0, stream>>>(y_f, y_b, m_f, m_b, ycat_bf);

    // 5) projection 1024 -> 512 via WMMA (+bias)
    k_cvt_bf16<<<cdiv(DI * 2 * DI, 256), 256, 0, stream>>>(W_pro_to, Wpro_bf, DI * 2 * DI);
    gemm(ycat_bf, Wpro_bf, y2, b_pro_to, NROW, DI, 2 * DI);

    // 6) center mask, apply + convert
    k_rowdist<<<NROW, 128, 0, stream>>>(y2, dv, refC);
    k_mask<<<B_, 256, 0, stream>>>(dv, m_c, (float)refC, sigCenter);
    k_apply_mask_cvt<<<cdiv(NROW * DI, 256), 256, 0, stream>>>(y2, m_c, y2m_bf);

    // 7) token attention: logits & VV via WMMA, softmax, then T = Atok@VV via WMMA
    k_cvt_bf16<<<cdiv(S_ * DI, 256), 256, 0, stream>>>(token_wA, wA_bf, S_ * DI);
    k_cvt_t_bf16<<<cdiv(DI * DI, 256), 256, 0, stream>>>(token_wV, wVT_bf, DI, DI);
    gemm(y2m_bf, wA_bf, logits, nullptr, NROW, S_, DI);
    gemm(y2m_bf, wVT_bf, VV, nullptr, NROW, DI, DI);
    k_softmax<<<B_ * S_, 256, 0, stream>>>(logits, Atok);
    k_cvt_atok<<<cdiv(B_ * S_ * KP_, 256), 256, 0, stream>>>(Atok, Atok_bf);
    k_cvt_vvt<<<cdiv(B_ * DI * KP_, 256), 256, 0, stream>>>(VV, VVT_bf);
    for (int b = 0; b < B_; ++b) {
        gemm(Atok_bf + (size_t)b * S_ * KP_,
             VVT_bf  + (size_t)b * DI * KP_,
             Tb      + (size_t)b * S_ * DI,
             nullptr, S_, DI, KP_);
    }

    // 8) pooled-z gate and output projection
    k_pool<<<cdiv(B_ * S_ * DI, 256), 256, 0, stream>>>(xi_z, zp);
    k_out<<<cdiv(B_ * S_ * D_, 256), 256, 0, stream>>>(Tb, zp, W_out, out);
}
